// DisentangledCrossAttention_29609504539061
// MI455X (gfx1250) — matrix operand order repaired and checked by reading, not compiled, not run on hardware
//
#include <hip/hip_runtime.h>
#include <hip/hip_bf16.h>

// ---------------- problem constants ----------------
#define BB   2
#define TQ   1024
#define TKV  1024
#define DD   1024
#define HH   16
#define HD   64
#define SPAN 32          // buckets = 32, table width 2*SPAN = 64
#define MAXREL 128
#define KCH  128         // GEMM K-chunk staged in LDS

typedef __bf16 v16bf __attribute__((ext_vector_type(16)));
typedef __bf16 v8bf  __attribute__((ext_vector_type(8)));
typedef float  v8f   __attribute__((ext_vector_type(8)));

static __device__ inline v16bf cat8(v8bf lo, v8bf hi) {
  v16bf r;
#pragma unroll
  for (int i = 0; i < 8; ++i) { r[i] = lo[i]; r[i + 8] = hi[i]; }
  return r;
}

static __device__ inline v8f wmma_bf16(v16bf a, v16bf b, v8f c) {
  // D = A(16x32) * B(32x16) + C, f32 accumulate
  return __builtin_amdgcn_wmma_f32_16x16x32_bf16(false, a, false, b, (short)0, c,
                                                 false, false);
}

// A-fragment (16x32, M x K) from row-major src (ldK elements per row).
static __device__ inline v16bf load_a_frag(const __bf16* src, int ldK, int k0,
                                           int lane) {
  const int row = lane & 15, half = lane >> 4;
  const __bf16* p = src + (size_t)row * ldK + k0 + half * 16;
  return cat8(*(const v8bf*)p, *(const v8bf*)(p + 8));
}

// B-fragment (32x16, K x N) where source holds N rows of contiguous K.
static __device__ inline v16bf load_b_frag(const __bf16* src, int ldK, int k0,
                                           int lane) {
  const int col = lane & 15, half = lane >> 4;
  return *(const v16bf*)(src + (size_t)col * ldK + k0 + half * 16);
}

// ---------------- TDM (Tensor Data Mover) tile load -----------------------
#if __has_builtin(__builtin_amdgcn_tensor_load_to_lds)
#define USE_TDM 1
typedef unsigned int u32x4 __attribute__((ext_vector_type(4)));
typedef int i32x4 __attribute__((ext_vector_type(4)));
typedef int i32x8 __attribute__((ext_vector_type(8)));

// DMA a 64-row x KCH-element bf16 tile (row stride = rowStrideElems) from
// global into LDS at byte offset ldsOff (rows packed contiguously, ld=KCH).
// Descriptor per CDNA5 ISA; 8-byte units.
// This toolchain's builtin is the 6-arg (clang-23) form.
static __device__ inline void tdm_load_tile(const __bf16* gsrc, unsigned ldsOff,
                                            int rowStrideElems) {
  unsigned long long ga = (unsigned long long)(size_t)gsrc;
  u32x4 g0;
  g0[0] = 1u;                                        // count=1, no gather
  g0[1] = ldsOff;                                    // lds_addr
  g0[2] = (unsigned)(ga & 0xFFFFFFFFu);              // global_addr[31:0]
  g0[3] = (unsigned)((ga >> 32) & 0x01FFFFFFu) | (2u << 30);  // addr[56:32], type=2
  const unsigned dim0 = 1u << 20;                    // tensor dims: large (no OOB)
  const unsigned dim1 = 1u << 20;
  const unsigned st0  = (unsigned)(rowStrideElems / 4);  // elems*2B / 8B units
  const unsigned td0  = KCH / 4;                     // tile_dim0: 32 x 8B
  const unsigned td1  = 64;                          // tile_dim1: 64 rows
  i32x8 g1;
  g1[0] = (int)(3u << 16);                           // mask=0, data_size=8B
  g1[1] = (int)((dim0 & 0xFFFFu) << 16);             // dim0[15:0] << 16
  g1[2] = (int)((dim0 >> 16) | ((dim1 & 0xFFFFu) << 16));
  g1[3] = (int)((dim1 >> 16) | (td0 << 16));         // dim1[31:16] | tile_dim0
  g1[4] = (int)td1;                                  // tile_dim1 | tile_dim2=0
  g1[5] = (int)st0;                                  // dim0_stride[31:0]
  g1[6] = 0;                                         // stride hi / dim1_stride
  g1[7] = 0;
  i32x4 z4 = {0, 0, 0, 0};                           // groups 2/3 unused (2-D tile)
  i32x8 z8 = {0, 0, 0, 0, 0, 0, 0, 0};               // extra operand (clang-23 form)
  __builtin_amdgcn_tensor_load_to_lds(g0, g1, z4, z4, z8, 0);
}
#endif

// ---------------- utility kernels ----------------
__global__ void cvt_bf16_kernel(const float* __restrict__ x,
                                __bf16* __restrict__ y, int n) {
  int i = blockIdx.x * blockDim.x + threadIdx.x;
  int stride = gridDim.x * blockDim.x;
  for (; i < n; i += stride) y[i] = (__bf16)x[i];
}

__global__ __launch_bounds__(256) void ln_kernel(const float* __restrict__ x,
                                                 const float* __restrict__ g,
                                                 const float* __restrict__ b,
                                                 __bf16* __restrict__ out) {
  __shared__ float red[256];
  const int row = blockIdx.x, tid = threadIdx.x;
  const float* xr = x + (size_t)row * DD;
  float s = 0.f;
  for (int i = tid; i < DD; i += 256) s += xr[i];
  red[tid] = s; __syncthreads();
  for (int st = 128; st; st >>= 1) { if (tid < st) red[tid] += red[tid + st]; __syncthreads(); }
  const float mu = red[0] / (float)DD;
  __syncthreads();
  float v = 0.f;
  for (int i = tid; i < DD; i += 256) { float d = xr[i] - mu; v += d * d; }
  red[tid] = v; __syncthreads();
  for (int st = 128; st; st >>= 1) { if (tid < st) red[tid] += red[tid + st]; __syncthreads(); }
  const float inv = rsqrtf(red[0] / (float)DD + 1e-5f);
  for (int i = tid; i < DD; i += 256)
    out[(size_t)row * DD + i] = (__bf16)((xr[i] - mu) * inv * g[i] + b[i]);
}

// log-bucket index tables
__global__ void bucket_kernel(unsigned char* __restrict__ tc,
                              unsigned char* __restrict__ tp) {
  int idx = blockIdx.x * blockDim.x + threadIdx.x;
  if (idx >= TQ * TKV) return;
  const int i = idx >> 10, j = idx & 1023;
  const int rel = i - j;
  const int mid = SPAN / 2;  // 16
  const int sign = (rel > 0) - (rel < 0);
  const int abs_pos = (rel < mid && rel > -mid) ? (mid - 1) : (rel < 0 ? -rel : rel);
  int bucket;
  if (abs_pos <= mid) bucket = rel;
  else {
    float log_pos = ceilf(logf((float)abs_pos / (float)mid) /
                          logf((float)(MAXREL - 1) / (float)mid) * (float)(mid - 1)) +
                    (float)mid;
    bucket = (int)log_pos * sign;
  }
  int c = bucket + SPAN;  c = c < 0 ? 0 : (c > 2 * SPAN - 1 ? 2 * SPAN - 1 : c);
  int p = -bucket + SPAN; p = p < 0 ? 0 : (p > 2 * SPAN - 1 ? 2 * SPAN - 1 : p);
  tc[idx] = (unsigned char)c;
  tp[idx] = (unsigned char)p;
}

// ------------- LDS-tiled WMMA GEMM: C = A(MxK) * W^T(NxK) + bias -------------
// 128 threads = 4 waves; workgroup tile 64x64; each wave a 32x32 block (2x2
// WMMA frags with A/B reuse). K staged in LDS in KCH chunks, double-buffered
// via the Tensor Data Mover when available.
// MODE 0: bf16 row-major (M,N)
// MODE 1: bf16 scattered to (B,H,T,HD)   [Q, K projections]
// MODE 2: bf16 scattered to (B,H,HD,T)   [V projection, pre-transposed]
// MODE 3: f32 row-major (M,N)            [final output]
template <int MODE>
__global__ __launch_bounds__(128) void gemm_tiled(const __bf16* __restrict__ A,
                                                  const __bf16* __restrict__ W,
                                                  const float* __restrict__ bias,
                                                  void* __restrict__ Cout, int M,
                                                  int N, int K, int Trows) {
  __shared__ alignas(64) __bf16 lA[2][64 * KCH];
  __shared__ alignas(64) __bf16 lB[2][64 * KCH];
  const int lane = threadIdx.x & 31;
  const int wave = threadIdx.x >> 5;
  const int half = lane >> 4, lcol = lane & 15;
  const int nt = blockIdx.x, mt = blockIdx.y;       // 64-wide tiles
  const int mi = (wave >> 1) * 32, ni = (wave & 1) * 32;
  const int NC = K / KCH;

  v8f acc[2][2];
#pragma unroll
  for (int a = 0; a < 2; ++a)
#pragma unroll
    for (int b = 0; b < 2; ++b) acc[a][b] = v8f{0.f,0.f,0.f,0.f,0.f,0.f,0.f,0.f};

#if defined(USE_TDM)
  if (wave == 0) {
    tdm_load_tile(A + (size_t)(mt * 64) * K, (unsigned)(size_t)&lA[0][0], K);
    tdm_load_tile(W + (size_t)(nt * 64) * K, (unsigned)(size_t)&lB[0][0], K);
  }
#endif

  for (int c = 0; c < NC; ++c) {
#if defined(USE_TDM)
    if (wave == 0) {
      if (c + 1 < NC) {
        const int nb = (c + 1) & 1;
        tdm_load_tile(A + (size_t)(mt * 64) * K + (c + 1) * KCH,
                      (unsigned)(size_t)&lA[nb][0], K);
        tdm_load_tile(W + (size_t)(nt * 64) * K + (c + 1) * KCH,
                      (unsigned)(size_t)&lB[nb][0], K);
        __builtin_amdgcn_s_wait_tensorcnt(2);  // chunk c landed
      } else {
        __builtin_amdgcn_s_wait_tensorcnt(0);
      }
    }
    __syncthreads();
    const __bf16* As = &lA[c & 1][0];
    const __bf16* Bs = &lB[c & 1][0];
#else
    // cooperative copy fallback
    for (int t = threadIdx.x; t < 64 * (KCH / 8); t += 128) {
      const int row = t / (KCH / 8), seg = t % (KCH / 8);
      *(v8bf*)&lA[0][row * KCH + seg * 8] =
          *(const v8bf*)(A + (size_t)(mt * 64 + row) * K + c * KCH + seg * 8);
      *(v8bf*)&lB[0][row * KCH + seg * 8] =
          *(const v8bf*)(W + (size_t)(nt * 64 + row) * K + c * KCH + seg * 8);
    }
    __syncthreads();
    const __bf16* As = &lA[0][0];
    const __bf16* Bs = &lB[0][0];
#endif
#pragma unroll
    for (int kk = 0; kk < KCH; kk += 32) {
      v16bf a0 = load_a_frag(As + (size_t)mi * KCH, KCH, kk, lane);
      v16bf a1 = load_a_frag(As + (size_t)(mi + 16) * KCH, KCH, kk, lane);
      v16bf b0 = load_b_frag(Bs + (size_t)ni * KCH, KCH, kk, lane);
      v16bf b1 = load_b_frag(Bs + (size_t)(ni + 16) * KCH, KCH, kk, lane);
      acc[0][0] = wmma_bf16(a0, b0, acc[0][0]);
      acc[0][1] = wmma_bf16(a0, b1, acc[0][1]);
      acc[1][0] = wmma_bf16(a1, b0, acc[1][0]);
      acc[1][1] = wmma_bf16(a1, b1, acc[1][1]);
    }
    __syncthreads();
  }

#pragma unroll
  for (int sm = 0; sm < 2; ++sm)
#pragma unroll
    for (int sn = 0; sn < 2; ++sn) {
      const int n = nt * 64 + ni + sn * 16 + lcol;
      const float bv = bias[n];
#pragma unroll
      for (int r = 0; r < 8; ++r) {
        const int m = mt * 64 + mi + sm * 16 + r + half * 8;
        const float v = acc[sm][sn][r] + bv;
        if (MODE == 3) {
          ((float*)Cout)[(size_t)m * N + n] = v;
        } else {
          const __bf16 o = (__bf16)v;
          if (MODE == 0) {
            ((__bf16*)Cout)[(size_t)m * N + n] = o;
          } else {
            const int bidx = m / Trows, t = m % Trows;
            const int hh = n >> 6, hd = n & 63;
            if (MODE == 1)
              ((__bf16*)Cout)[(((size_t)bidx * HH + hh) * Trows + t) * HD + hd] = o;
            else
              ((__bf16*)Cout)[(((size_t)bidx * HH + hh) * HD + hd) * Trows + t] = o;
          }
        }
      }
    }
}

// ---------------- rel-position projections: out(bh,t,64) = X(bh,t,64) @ pos^T ----
__global__ __launch_bounds__(32) void relproj_wmma(const __bf16* __restrict__ X,
                                                   const __bf16* __restrict__ pos,
                                                   float* __restrict__ out) {
  const int lane = threadIdx.x & 31;
  const int half = lane >> 4, lcol = lane & 15;
  const int pt = blockIdx.x, qt = blockIdx.y, bh = blockIdx.z;
  const int h = bh & (HH - 1);
  const __bf16* ap = X + ((size_t)bh * TQ + qt * 16) * HD;
  const __bf16* pp = pos + (size_t)(pt * 16) * DD + h * HD;  // ldK = DD
  v8f acc = {0.f, 0.f, 0.f, 0.f, 0.f, 0.f, 0.f, 0.f};
#pragma unroll
  for (int k0 = 0; k0 < HD; k0 += 32) {
    v16bf af = load_a_frag(ap, HD, k0, lane);
    v16bf bf = load_b_frag(pp, DD, k0, lane);
    acc = wmma_bf16(af, bf, acc);
  }
#pragma unroll
  for (int r = 0; r < 8; ++r) {
    const int q = qt * 16 + r + half * 8;
    out[((size_t)bh * TQ + q) * (2 * SPAN) + pt * 16 + lcol] = acc[r];
  }
}

// ---------------- fused attention (flash-style, one wave per 16-q tile) --------
__global__ __launch_bounds__(32) void attn_wmma(
    const __bf16* __restrict__ Q, const __bf16* __restrict__ K,
    const __bf16* __restrict__ Vt, const float* __restrict__ c2p,
    const float* __restrict__ p2c, const unsigned char* __restrict__ tc,
    const unsigned char* __restrict__ tp, const unsigned char* __restrict__ mask,
    __bf16* __restrict__ ctx) {
  __shared__ __bf16 pld[16 * 32];  // P-tile staging for C->A fragment transpose
  const int lane = threadIdx.x & 31;
  const int half = lane >> 4, lcol = lane & 15;
  const int qtile = blockIdx.x, bh = blockIdx.y;
  const int b = bh >> 4, h = bh & (HH - 1);
  const int qbase = qtile * 16;

  const float inv_scale  = 0.07216878364870323f;  // 1/sqrt(3*HD)
  const float inv_pscale = 0.08838834764831845f;  // 1/sqrt(2*HD)

  const __bf16* Qb = Q + ((size_t)bh * TQ + qbase) * HD;
  v16bf qf0 = load_a_frag(Qb, HD, 0, lane);
  v16bf qf1 = load_a_frag(Qb, HD, 32, lane);

  v8f o0 = {0.f,0.f,0.f,0.f,0.f,0.f,0.f,0.f};
  v8f o1 = o0, o2 = o0, o3 = o0;
  float mr[8], lr[8];
#pragma unroll
  for (int r = 0; r < 8; ++r) { mr[r] = -1e30f; lr[r] = 0.f; }

  const __bf16* Kb = K + (size_t)bh * TKV * HD;
  const __bf16* Vb = Vt + (size_t)bh * HD * TKV;

  for (int kv0 = 0; kv0 < TKV; kv0 += 32) {
    v8f s[2];
#pragma unroll
    for (int sub = 0; sub < 2; ++sub) {
      const int kvc = kv0 + sub * 16 + lcol;
      const __bf16* kp = Kb + (size_t)kvc * HD;
      v16bf b0 = *(const v16bf*)(kp + half * 16);
      v16bf b1 = *(const v16bf*)(kp + 32 + half * 16);
      v8f a2 = {0.f,0.f,0.f,0.f,0.f,0.f,0.f,0.f};
      a2 = wmma_bf16(qf0, b0, a2);
      a2 = wmma_bf16(qf1, b1, a2);
      const unsigned char mk = mask[b * TKV + kvc];
#pragma unroll
      for (int r = 0; r < 8; ++r) {
        const int qrow = qbase + r + half * 8;
        const int ic = tc[(size_t)qrow * TKV + kvc];
        const int ip = tp[(size_t)kvc * TQ + qrow];
        const float bias =
            c2p[((size_t)bh * TQ + qrow) * (2 * SPAN) + ic] +
            p2c[((size_t)bh * TKV + kvc) * (2 * SPAN) + ip];
        const float v = a2[r] * inv_scale + bias * inv_pscale;
        a2[r] = mk ? -1e30f : v;
      }
      s[sub] = a2;
    }
#pragma unroll
    for (int r = 0; r < 8; ++r) {
      float v = fmaxf(s[0][r], s[1][r]);
#pragma unroll
      for (int off = 8; off; off >>= 1) v = fmaxf(v, __shfl_xor(v, off));
      const float mnew = fmaxf(mr[r], v);
      const float f = __expf(mr[r] - mnew);
      const float p0 = __expf(s[0][r] - mnew);
      const float p1 = __expf(s[1][r] - mnew);
      float rs = p0 + p1;
#pragma unroll
      for (int off = 8; off; off >>= 1) rs += __shfl_xor(rs, off);
      lr[r] = lr[r] * f + rs;
      mr[r] = mnew;
      o0[r] *= f; o1[r] *= f; o2[r] *= f; o3[r] *= f;
      s[0][r] = p0; s[1][r] = p1;
    }
#pragma unroll
    for (int sub = 0; sub < 2; ++sub)
#pragma unroll
      for (int r = 0; r < 8; ++r)
        pld[(r + half * 8) * 32 + sub * 16 + lcol] = (__bf16)s[sub][r];
    asm volatile("s_wait_dscnt 0" ::: "memory");
    v16bf pf = load_a_frag(pld, 32, 0, lane);
    v16bf vf0 = *(const v16bf*)(Vb + (size_t)(0 * 16 + lcol) * TKV + kv0 + half * 16);
    v16bf vf1 = *(const v16bf*)(Vb + (size_t)(1 * 16 + lcol) * TKV + kv0 + half * 16);
    v16bf vf2 = *(const v16bf*)(Vb + (size_t)(2 * 16 + lcol) * TKV + kv0 + half * 16);
    v16bf vf3 = *(const v16bf*)(Vb + (size_t)(3 * 16 + lcol) * TKV + kv0 + half * 16);
    o0 = wmma_bf16(pf, vf0, o0);
    o1 = wmma_bf16(pf, vf1, o1);
    o2 = wmma_bf16(pf, vf2, o2);
    o3 = wmma_bf16(pf, vf3, o3);
  }
#pragma unroll
  for (int r = 0; r < 8; ++r) {
    const float inv = 1.0f / lr[r];
    const int qrow = qbase + r + half * 8;
    const size_t base = ((size_t)b * TQ + qrow) * DD + h * HD;
    ctx[base + 0 * 16 + lcol] = (__bf16)(o0[r] * inv);
    ctx[base + 1 * 16 + lcol] = (__bf16)(o1[r] * inv);
    ctx[base + 2 * 16 + lcol] = (__bf16)(o2[r] * inv);
    ctx[base + 3 * 16 + lcol] = (__bf16)(o3[r] * inv);
  }
}

// ---------------- host side ----------------
extern "C" void kernel_launch(void* const* d_in, const int* in_sizes, int n_in,
                              void* d_out, int out_size, void* d_ws,
                              size_t ws_size, hipStream_t stream) {
  const float* q_in  = (const float*)d_in[0];
  const float* kv_in = (const float*)d_in[1];
  const float* Wq = (const float*)d_in[2];  const float* bq = (const float*)d_in[3];
  const float* Wk = (const float*)d_in[4];  const float* bk = (const float*)d_in[5];
  const float* Wv = (const float*)d_in[6];  const float* bv = (const float*)d_in[7];
  const float* Wo = (const float*)d_in[8];  const float* bo = (const float*)d_in[9];
  const float* Wpk = (const float*)d_in[10]; const float* bpk = (const float*)d_in[11];
  const float* Wpq = (const float*)d_in[12]; const float* bpq = (const float*)d_in[13];
  const float* rel_emb = (const float*)d_in[14];
  const float* ln_g = (const float*)d_in[15];
  const float* ln_b = (const float*)d_in[16];
  const unsigned char* mask = (const unsigned char*)d_in[17];
  float* out = (float*)d_out;

  char* p = (char*)d_ws;
  auto alloc = [&](size_t bytes) -> char* {
    char* r = p;
    p += (bytes + 255) & ~(size_t)255;
    return r;
  };
  const size_t nAct = (size_t)BB * TQ * DD;
  const size_t nW = (size_t)DD * DD;
  __bf16* Xq   = (__bf16*)alloc(nAct * 2);
  __bf16* Xkv  = (__bf16*)alloc(nAct * 2);
  __bf16* Wqb  = (__bf16*)alloc(nW * 2);
  __bf16* Wkb  = (__bf16*)alloc(nW * 2);
  __bf16* Wvb  = (__bf16*)alloc(nW * 2);
  __bf16* Wob  = (__bf16*)alloc(nW * 2);
  __bf16* Wpkb = (__bf16*)alloc(nW * 2);
  __bf16* Wpqb = (__bf16*)alloc(nW * 2);
  __bf16* re   = (__bf16*)alloc((size_t)2 * SPAN * DD * 2);
  __bf16* Qb   = (__bf16*)alloc(nAct * 2);      // (B,H,TQ,HD)
  __bf16* Kb   = (__bf16*)alloc(nAct * 2);      // (B,H,TKV,HD)
  __bf16* Vt   = (__bf16*)alloc(nAct * 2);      // (B,H,HD,TKV)
  __bf16* posk = (__bf16*)alloc((size_t)2 * SPAN * DD * 2);
  __bf16* posq = (__bf16*)alloc((size_t)2 * SPAN * DD * 2);
  float*  c2p  = (float*)alloc((size_t)BB * HH * TQ * 2 * SPAN * 4);
  float*  p2c  = (float*)alloc((size_t)BB * HH * TKV * 2 * SPAN * 4);
  unsigned char* tc = (unsigned char*)alloc((size_t)TQ * TKV);
  unsigned char* tp = (unsigned char*)alloc((size_t)TQ * TKV);
  __bf16* ctx  = (__bf16*)alloc(nAct * 2);

  // 1) bf16 conversions
  hipLaunchKernelGGL(cvt_bf16_kernel, dim3(2048), dim3(256), 0, stream, q_in, Xq, (int)nAct);
  hipLaunchKernelGGL(cvt_bf16_kernel, dim3(2048), dim3(256), 0, stream, kv_in, Xkv, (int)nAct);
  hipLaunchKernelGGL(cvt_bf16_kernel, dim3(1024), dim3(256), 0, stream, Wq, Wqb, (int)nW);
  hipLaunchKernelGGL(cvt_bf16_kernel, dim3(1024), dim3(256), 0, stream, Wk, Wkb, (int)nW);
  hipLaunchKernelGGL(cvt_bf16_kernel, dim3(1024), dim3(256), 0, stream, Wv, Wvb, (int)nW);
  hipLaunchKernelGGL(cvt_bf16_kernel, dim3(1024), dim3(256), 0, stream, Wo, Wob, (int)nW);
  hipLaunchKernelGGL(cvt_bf16_kernel, dim3(1024), dim3(256), 0, stream, Wpk, Wpkb, (int)nW);
  hipLaunchKernelGGL(cvt_bf16_kernel, dim3(1024), dim3(256), 0, stream, Wpq, Wpqb, (int)nW);

  // 2) LayerNorm(rel_emb) -> bf16
  hipLaunchKernelGGL(ln_kernel, dim3(2 * SPAN), dim3(256), 0, stream, rel_emb, ln_g, ln_b, re);

  // 3) projections (LDS-tiled WMMA GEMMs, TDM double-buffered)
  const int M = BB * TQ, N = DD, Kd = DD;
  hipLaunchKernelGGL((gemm_tiled<1>), dim3(N / 64, M / 64), dim3(128), 0, stream,
                     Xq, Wqb, bq, (void*)Qb, M, N, Kd, TQ);
  hipLaunchKernelGGL((gemm_tiled<1>), dim3(N / 64, M / 64), dim3(128), 0, stream,
                     Xkv, Wkb, bk, (void*)Kb, M, N, Kd, TKV);
  hipLaunchKernelGGL((gemm_tiled<2>), dim3(N / 64, M / 64), dim3(128), 0, stream,
                     Xkv, Wvb, bv, (void*)Vt, M, N, Kd, TKV);
  hipLaunchKernelGGL((gemm_tiled<0>), dim3(N / 64, 1), dim3(128), 0, stream,
                     re, Wpkb, bpk, (void*)posk, 2 * SPAN, N, Kd, 1);
  hipLaunchKernelGGL((gemm_tiled<0>), dim3(N / 64, 1), dim3(128), 0, stream,
                     re, Wpqb, bpq, (void*)posq, 2 * SPAN, N, Kd, 1);

  // 4) bucket index tables
  hipLaunchKernelGGL(bucket_kernel, dim3((TQ * TKV) / 256), dim3(256), 0, stream, tc, tp);

  // 5) c2p / p2c rel projections (WMMA)
  hipLaunchKernelGGL(relproj_wmma, dim3((2 * SPAN) / 16, TQ / 16, BB * HH), dim3(32),
                     0, stream, Qb, posk, c2p);
  hipLaunchKernelGGL(relproj_wmma, dim3((2 * SPAN) / 16, TKV / 16, BB * HH), dim3(32),
                     0, stream, Kb, posq, p2c);

  // 6) fused attention
  hipLaunchKernelGGL(attn_wmma, dim3(TQ / 16, BB * HH), dim3(32), 0, stream, Qb, Kb,
                     Vt, c2p, p2c, tc, tp, mask, ctx);

  // 7) output projection -> f32
  hipLaunchKernelGGL((gemm_tiled<3>), dim3(N / 64, M / 64), dim3(128), 0, stream, ctx,
                     Wob, bo, (void*)out, M, N, Kd, 1);
}
